// CausalSelfAttention_78855599554756
// MI455X (gfx1250) — compile-verified, matrix operand order checked by reading, not run on hardware
//
#include <hip/hip_runtime.h>
#include <hip/hip_bf16.h>

// ---------------------------------------------------------------------------
// CDNA5 (gfx1250) causal self-attention, bf16 WMMA pipeline, wave32.
// WMMA shape 16x16x32 (bf16 in, f32 acc). Double-buffered GEMM k-loops so the
// compiler can use partial s_wait_loadcnt instead of full drains per WMMA.
// ---------------------------------------------------------------------------

typedef __attribute__((ext_vector_type(16))) __bf16         v16bf;
typedef __attribute__((ext_vector_type(8)))  float          v8f;
typedef __attribute__((ext_vector_type(4)))  unsigned int   u32x4;
typedef __attribute__((ext_vector_type(4)))  float          f32x4;
typedef __attribute__((ext_vector_type(4)))  unsigned short u16x4;

union FragB {
    v16bf v;
    u32x4 q[2];
};

__device__ __forceinline__ unsigned short f2bf(float f) {
    unsigned int u = __float_as_uint(f);
    u += 0x7FFFu + ((u >> 16) & 1u);          // round-to-nearest-even
    return (unsigned short)(u >> 16);
}
__device__ __forceinline__ float bf2f(unsigned short h) {
    return __uint_as_float(((unsigned int)h) << 16);
}

__device__ __forceinline__ v8f wmma_bf16(v16bf a, v16bf b, v8f c) {
    // D = A(16x32) * B(32x16) + C(16x16), f32 accumulate
    return __builtin_amdgcn_wmma_f32_16x16x32_bf16(
        false, a, false, b, (short)0, c, false, false);
}

// Load a 16x32 bf16 fragment (A-layout; also used for B loaded from B^T rows).
// ISA 7.12.2: lanes 0-15 hold row r0+lane with K = {0..7, 16..23},
//             lanes 16-31 hold row r0+lane-16 with K = {8..15, 24..31}.
__device__ __forceinline__ FragB load_frag(const unsigned short* __restrict__ p,
                                           int ld, int r0, int k0, int lane) {
    const int l16  = lane & 15;
    const int koff = (lane >> 4) << 3;        // 0 or 8
    const unsigned short* base = p + (size_t)(r0 + l16) * ld + k0 + koff;
    FragB f;
    f.q[0] = *(const u32x4*)(base);           // K = koff .. koff+7
    f.q[1] = *(const u32x4*)(base + 16);      // K = koff+16 .. koff+23
    return f;
}

// ---------------------------------------------------------------------------
// Kernel 1: fp32 -> bf16 elementwise convert (x)
// ---------------------------------------------------------------------------
__global__ void cvt_bf16_kernel(const float* __restrict__ in,
                                unsigned short* __restrict__ out, int n4) {
    int i = blockIdx.x * blockDim.x + threadIdx.x;
    if (i >= n4) return;
    f32x4 f = *(const f32x4*)(in + (size_t)i * 4);
    u16x4 o;
    o[0] = f2bf(f[0]); o[1] = f2bf(f[1]); o[2] = f2bf(f[2]); o[3] = f2bf(f[3]);
    *(u16x4*)(out + (size_t)i * 4) = o;
}

// ---------------------------------------------------------------------------
// Kernel 2: LDS-tiled transpose + convert: in[R x C] f32 -> out[C x R] bf16
// ---------------------------------------------------------------------------
__global__ void transpose_bf16_kernel(const float* __restrict__ in,
                                      unsigned short* __restrict__ out,
                                      int R, int C) {
    __shared__ float tile[32][33];
    const int c0 = blockIdx.x * 32;
    const int r0 = blockIdx.y * 32;
    const int tx = threadIdx.x;               // 0..31
    const int ty = threadIdx.y;               // 0..7
#pragma unroll
    for (int i = 0; i < 32; i += 8)
        tile[ty + i][tx] = in[(size_t)(r0 + ty + i) * C + (c0 + tx)];
    __syncthreads();
#pragma unroll
    for (int i = 0; i < 32; i += 8)
        out[(size_t)(c0 + ty + i) * R + (r0 + tx)] = f2bf(tile[tx][ty + i]);
}

// ---------------------------------------------------------------------------
// GEMM core: wave tile 32(M) x 64(N), double-buffered k-loop (step 32).
// A from Abf[M x 2048] rows, B from Bt[N x 2048] rows (B^T storage).
// acc[i][t]: i = M subtile (2 x 16), t = N subtile (4 x 16).
// ---------------------------------------------------------------------------
__device__ __forceinline__ void gemm_load_tiles(
    const unsigned short* __restrict__ A, const unsigned short* __restrict__ Bt,
    int mtile, int ntile, int k, int lane, FragB a[2], FragB b[4]) {
    a[0] = load_frag(A, 2048, mtile,      k, lane);
    a[1] = load_frag(A, 2048, mtile + 16, k, lane);
#pragma unroll
    for (int t = 0; t < 4; t++)
        b[t] = load_frag(Bt, 2048, ntile + t * 16, k, lane);
}

__device__ __forceinline__ void gemm_mma8(v8f acc[2][4], FragB a[2], FragB b[4]) {
#pragma unroll
    for (int i = 0; i < 2; i++)
#pragma unroll
        for (int t = 0; t < 4; t++)
            acc[i][t] = wmma_bf16(a[i].v, b[t].v, acc[i][t]);
}

// ---------------------------------------------------------------------------
// Kernel 3: QKV GEMM.  [4096 x 6144] = Xbf[4096 x 2048] * WqkvT[6144 x 2048]^T
// 4 waves arranged 2x2 -> WG tile 64(M) x 128(N).
// Epilogue scatters into Q[bh][n][d], K[bh][n][d], V^T[bh][d][n] (bf16).
// ---------------------------------------------------------------------------
__global__ __launch_bounds__(128) void qkv_gemm_kernel(
    const unsigned short* __restrict__ Xbf,
    const unsigned short* __restrict__ WqkvT,
    unsigned short* __restrict__ Qr,
    unsigned short* __restrict__ Kr,
    unsigned short* __restrict__ Vt) {
    const int lane  = threadIdx.x & 31;
    const int wave  = threadIdx.x >> 5;
    const int wm    = wave >> 1, wn = wave & 1;
    const int mtile = blockIdx.x * 64 + wm * 32;
    const int ntile = blockIdx.y * 128 + wn * 64;
    const int l16   = lane & 15;
    const int half  = (lane >> 4) << 3;

    v8f acc[2][4] = {};
    FragB a[2], b[4];
    gemm_load_tiles(Xbf, WqkvT, mtile, ntile, 0, lane, a, b);
    for (int k = 32; k < 2048; k += 32) {
        FragB an[2], bn[4];
        gemm_load_tiles(Xbf, WqkvT, mtile, ntile, k, lane, an, bn);
        __builtin_prefetch(Xbf + (size_t)(mtile + l16) * 2048 + k + 32, 0, 1);
        gemm_mma8(acc, a, b);
#pragma unroll
        for (int i = 0; i < 2; i++) a[i] = an[i];
#pragma unroll
        for (int t = 0; t < 4; t++) b[t] = bn[t];
    }
    gemm_mma8(acc, a, b);

    // C layout: vgpr r -> row mtile+i*16+half+r ; lane -> col ntile+t*16+l16
#pragma unroll
    for (int i = 0; i < 2; i++) {
#pragma unroll
        for (int t = 0; t < 4; t++) {
            const int c     = ntile + t * 16 + l16;   // 0..6143
            const int which = c >> 11;                // 0=q 1=k 2=v
            const int head  = (c >> 7) & 15;
            const int dd    = c & 127;
#pragma unroll
            for (int r = 0; r < 8; r++) {
                const int m  = mtile + i * 16 + half + r;   // 0..4095
                const int bb = m >> 11;
                const int n  = m & 2047;
                const int bh = bb * 16 + head;
                const unsigned short val = f2bf(acc[i][t][r]);
                if (which == 0)
                    Qr[((size_t)bh * 2048 + n) * 128 + dd] = val;
                else if (which == 1)
                    Kr[((size_t)bh * 2048 + n) * 128 + dd] = val;
                else
                    Vt[((size_t)bh * 128 + dd) * 2048 + n] = val;  // V^T
            }
        }
    }
}

// ---------------------------------------------------------------------------
// Kernel 4: RoPE in place on Q and K, [32][2048][128] bf16, f32 math.
// ---------------------------------------------------------------------------
__global__ void rope_kernel(unsigned short* __restrict__ Qr,
                            unsigned short* __restrict__ Kr) {
    const int idx = blockIdx.x * blockDim.x + threadIdx.x;  // 32*2048*64 pairs
    const int i   = idx & 63;                               // dim-pair index
    const int n   = (idx >> 6) & 2047;                      // position
    const int bh  = idx >> 17;
    if (bh >= 32) return;
    // theta_i = 10000^(-2i/128) = exp(-2i/128 * ln(10000))
    const float theta = __expf(-(float)(2 * i) * (9.210340371976184f / 128.0f));
    const float ang   = (float)n * theta;
    float s, c;
    __sincosf(ang, &s, &c);
    const size_t base = ((size_t)bh * 2048 + n) * 128 + 2 * i;
    {
        const float xe = bf2f(Qr[base]), xo = bf2f(Qr[base + 1]);
        Qr[base]     = f2bf(xe * c - xo * s);
        Qr[base + 1] = f2bf(xe * s + xo * c);
    }
    {
        const float xe = bf2f(Kr[base]), xo = bf2f(Kr[base + 1]);
        Kr[base]     = f2bf(xe * c - xo * s);
        Kr[base + 1] = f2bf(xe * s + xo * c);
    }
}

// ---------------------------------------------------------------------------
// Kernel 5: causal flash attention.  8 waves/WG, each wave owns 16 query rows.
// S = Q*K^T (8 WMMA per 32-key tile, all K frags loaded as one clause),
// online softmax with width-16 shuffles, P via 1KB/wave LDS tile
// (C-layout -> A-layout), O += P*V (8 WMMA, V frags double-buffered).
// ---------------------------------------------------------------------------
__global__ __launch_bounds__(256) void flash_kernel(
    const unsigned short* __restrict__ Qr,
    const unsigned short* __restrict__ Kr,
    const unsigned short* __restrict__ Vt,
    unsigned short* __restrict__ Ybf) {
    __shared__ unsigned short Plds[8][16 * 32];   // per-wave P tile (bf16)

    const int lane = threadIdx.x & 31;
    const int wave = threadIdx.x >> 5;
    const int l16  = lane & 15;
    const int koff = (lane >> 4) << 3;            // 0 or 8
    const int half = koff;                        // row offset of this half-wave
    const int bh   = blockIdx.x;                  // 0..31
    const int q0   = blockIdx.y * 128 + wave * 16;

    const unsigned short* Qp = Qr + (size_t)bh * 2048 * 128;
    const unsigned short* Kp = Kr + (size_t)bh * 2048 * 128;
    const unsigned short* Vp = Vt + (size_t)bh * 128 * 2048;

    FragB qf[4];
#pragma unroll
    for (int kc = 0; kc < 4; kc++)
        qf[kc] = load_frag(Qp, 128, q0, kc * 32, lane);

    v8f o[8] = {};
    float mrow[8], lrow[8];
#pragma unroll
    for (int r = 0; r < 8; r++) { mrow[r] = -1e30f; lrow[r] = 0.0f; }

    const float scale = 0.08838834764831845f;     // 1/sqrt(128)
    const int nKT = (q0 + 47) >> 5;               // ceil((q0+16)/32)

    for (int kt = 0; kt < nKT; kt++) {
        const int kb = kt * 32;
        // --- S = Q K^T : load all 8 K fragments first (distinct regs) ---
        FragB kf[8];
#pragma unroll
        for (int kc = 0; kc < 4; kc++) {
            kf[kc * 2]     = load_frag(Kp, 128, kb,      kc * 32, lane);
            kf[kc * 2 + 1] = load_frag(Kp, 128, kb + 16, kc * 32, lane);
        }
        v8f s0 = {}, s1 = {};
#pragma unroll
        for (int kc = 0; kc < 4; kc++) {
            s0 = wmma_bf16(qf[kc].v, kf[kc * 2].v,     s0);
            s1 = wmma_bf16(qf[kc].v, kf[kc * 2 + 1].v, s1);
        }
        // --- online softmax ---
        const int key0 = kb + l16;
        const int key1 = key0 + 16;
#pragma unroll
        for (int r = 0; r < 8; r++) {
            const int mg = q0 + half + r;                     // query position
            float v0 = (key0 <= mg) ? s0[r] * scale : -1e30f; // causal mask
            float v1 = (key1 <= mg) ? s1[r] * scale : -1e30f;
            float t = fmaxf(v0, v1);
#pragma unroll
            for (int off = 1; off < 16; off <<= 1)
                t = fmaxf(t, __shfl_xor(t, off, 16));         // row max (16 cols)
            const float mn = fmaxf(mrow[r], t);
            const float alpha = __expf(mrow[r] - mn);
            const float p0 = __expf(v0 - mn);
            const float p1 = __expf(v1 - mn);
            float rs = p0 + p1;
#pragma unroll
            for (int off = 1; off < 16; off <<= 1)
                rs += __shfl_xor(rs, off, 16);                // row sum
            lrow[r] = lrow[r] * alpha + rs;
            mrow[r] = mn;
            Plds[wave][(half + r) * 32 + l16]      = f2bf(p0);
            Plds[wave][(half + r) * 32 + 16 + l16] = f2bf(p1);
#pragma unroll
            for (int dt = 0; dt < 8; dt++) o[dt][r] *= alpha;
        }
        // --- P back as A-fragment from LDS ---
        FragB pf;
        {
            const unsigned short* pb = &Plds[wave][l16 * 32 + koff];
            pf.q[0] = *(const u32x4*)(pb);
            pf.q[1] = *(const u32x4*)(pb + 16);
        }
        // --- O += P * V, V fragments double-buffered ---
        FragB vf0 = load_frag(Vp, 2048, 0, kb, lane);
#pragma unroll
        for (int dt = 0; dt < 8; dt++) {
            FragB vf1;
            if (dt < 7) vf1 = load_frag(Vp, 2048, (dt + 1) * 16, kb, lane);
            o[dt] = wmma_bf16(pf.v, vf0.v, o[dt]);
            if (dt < 7) vf0 = vf1;
        }
    }
    // epilogue: y[b, n, head*128 + d] = o / l
    const int b    = bh >> 4;
    const int head = bh & 15;
#pragma unroll
    for (int r = 0; r < 8; r++) {
        const float inv = 1.0f / lrow[r];
        const size_t row = (size_t)(b * 2048 + q0 + half + r) * 2048;
#pragma unroll
        for (int dt = 0; dt < 8; dt++)
            Ybf[row + head * 128 + dt * 16 + l16] = f2bf(o[dt][r] * inv);
    }
}

// ---------------------------------------------------------------------------
// Kernel 6: output projection. out[4096 x 2048] f32 = Ybf * WoT^T
// Same 32x64 wave tile, double-buffered k-loop.
// ---------------------------------------------------------------------------
__global__ __launch_bounds__(128) void out_gemm_kernel(
    const unsigned short* __restrict__ Ybf,
    const unsigned short* __restrict__ WoT,
    float* __restrict__ out) {
    const int lane  = threadIdx.x & 31;
    const int wave  = threadIdx.x >> 5;
    const int wm    = wave >> 1, wn = wave & 1;
    const int mtile = blockIdx.x * 64 + wm * 32;
    const int ntile = blockIdx.y * 128 + wn * 64;
    const int l16   = lane & 15;
    const int half  = (lane >> 4) << 3;

    v8f acc[2][4] = {};
    FragB a[2], b[4];
    gemm_load_tiles(Ybf, WoT, mtile, ntile, 0, lane, a, b);
    for (int k = 32; k < 2048; k += 32) {
        FragB an[2], bn[4];
        gemm_load_tiles(Ybf, WoT, mtile, ntile, k, lane, an, bn);
        __builtin_prefetch(Ybf + (size_t)(mtile + l16) * 2048 + k + 32, 0, 1);
        gemm_mma8(acc, a, b);
#pragma unroll
        for (int i = 0; i < 2; i++) a[i] = an[i];
#pragma unroll
        for (int t = 0; t < 4; t++) b[t] = bn[t];
    }
    gemm_mma8(acc, a, b);

#pragma unroll
    for (int i = 0; i < 2; i++)
#pragma unroll
        for (int t = 0; t < 4; t++) {
            const int c = ntile + t * 16 + l16;
#pragma unroll
            for (int r = 0; r < 8; r++)
                out[(size_t)(mtile + i * 16 + half + r) * 2048 + c] = acc[i][t][r];
        }
}

// ---------------------------------------------------------------------------
// Host launcher
// ---------------------------------------------------------------------------
extern "C" void kernel_launch(void* const* d_in, const int* in_sizes, int n_in,
                              void* d_out, int out_size, void* d_ws, size_t ws_size,
                              hipStream_t stream) {
    (void)in_sizes; (void)n_in; (void)out_size; (void)ws_size;
    const float* x    = (const float*)d_in[0];   // [2, 2048, 2048]
    const float* Wqkv = (const float*)d_in[1];   // [2048, 6144]
    const float* Wo   = (const float*)d_in[2];   // [2048, 2048]
    float* out        = (float*)d_out;           // [2, 2048, 2048]

    char* ws = (char*)d_ws;
    const size_t MiB = 1024 * 1024;
    unsigned short* Xbf   = (unsigned short*)(ws);             // 16 MiB
    unsigned short* WqkvT = (unsigned short*)(ws + 16 * MiB);  // 24 MiB
    unsigned short* WoT   = (unsigned short*)(ws + 40 * MiB);  //  8 MiB
    unsigned short* Qr    = (unsigned short*)(ws + 48 * MiB);  // 16 MiB
    unsigned short* Kr    = (unsigned short*)(ws + 64 * MiB);  // 16 MiB
    unsigned short* Vt    = (unsigned short*)(ws + 80 * MiB);  // 16 MiB
    unsigned short* Ybf   = (unsigned short*)(ws + 96 * MiB);  // 16 MiB (112 total)

    // 1) x -> bf16
    cvt_bf16_kernel<<<8192, 256, 0, stream>>>(x, Xbf, 2097152);
    // 2) weight transpose+convert
    dim3 tb(32, 8);
    transpose_bf16_kernel<<<dim3(192, 64), tb, 0, stream>>>(Wqkv, WqkvT, 2048, 6144);
    transpose_bf16_kernel<<<dim3(64, 64),  tb, 0, stream>>>(Wo,   WoT,   2048, 2048);
    // 3) QKV projection + scatter to per-head layouts
    qkv_gemm_kernel<<<dim3(64, 48), 128, 0, stream>>>(Xbf, WqkvT, Qr, Kr, Vt);
    // 4) RoPE on Q, K
    rope_kernel<<<16384, 256, 0, stream>>>(Qr, Kr);
    // 5) causal flash attention
    flash_kernel<<<dim3(32, 16), 256, 0, stream>>>(Qr, Kr, Vt, Ybf);
    // 6) output projection (f32 result)
    out_gemm_kernel<<<dim3(64, 16), 128, 0, stream>>>(Ybf, WoT, out);
}
